// CoarseMatching_91147795956266
// MI455X (gfx1250) — matrix operand order repaired and checked by reading, not compiled, not run on hardware
//
#include <hip/hip_runtime.h>
#include <hip/hip_bf16.h>

#define L_TOT 6400
#define C_DIM 256
#define LSIDE 80

typedef __bf16 bf16;
typedef bf16 v16bf __attribute__((ext_vector_type(16)));
typedef bf16 v8bf  __attribute__((ext_vector_type(8)));
typedef float v8f  __attribute__((ext_vector_type(8)));

union V16 { v16bf v; v8bf h[2]; };

// ---------------------------------------------------------------------------
// Pass 0: scale by C^-0.5, split fp32 -> bf16 hi + bf16 lo, row norms.
// One thread per row (rows 0..L-1 -> f0, L..2L-1 -> f1).
// ---------------------------------------------------------------------------
__global__ void prep_kernel(const float* __restrict__ f0, const float* __restrict__ f1,
                            bf16* __restrict__ f0h, bf16* __restrict__ f0l,
                            bf16* __restrict__ f1h, bf16* __restrict__ f1l,
                            float* __restrict__ n0, float* __restrict__ n1) {
  int r = blockIdx.x * blockDim.x + threadIdx.x;
  if (r >= 2 * L_TOT) return;
  const float* src; bf16 *dh, *dl; float* nrm; int row;
  if (r < L_TOT) { src = f0; dh = f0h; dl = f0l; nrm = n0; row = r; }
  else           { src = f1; dh = f1h; dl = f1l; nrm = n1; row = r - L_TOT; }
  const float scale = 0.0625f;  // 256^-0.5
  size_t base = (size_t)row * C_DIM;
  float s = 0.f;
  for (int k = 0; k < C_DIM; ++k) {
    float x = src[base + k] * scale;
    s += x * x;
    __hip_bfloat16 h = __float2bfloat16(x);
    float res = x - __bfloat162float(h);
    __hip_bfloat16 l = __float2bfloat16(res);
    dh[base + k] = *(bf16*)&h;
    dl[base + k] = *(bf16*)&l;
  }
  nrm[row] = s;
}

// ---------------------------------------------------------------------------
// Fused split-bf16 WMMA GEMM + streaming top-3 per row of
//   d[i][j] = nA[i] + nB[j] - 2 * (A @ B^T)[i][j]
// Block = 64 threads = 2 waves; each wave owns a 16-row tile, loops all 400
// 16-column tiles, K in 8 chunks of 32. A rows staged in LDS; B fragments
// loaded directly from global (L2 resident). S tile transposed via LDS so
// lanes 0..15 maintain per-row top-3 in registers.
// ---------------------------------------------------------------------------
__global__ __launch_bounds__(64)
void gemm_top3_kernel(const bf16* __restrict__ Ah, const bf16* __restrict__ Al,
                      const bf16* __restrict__ Bh, const bf16* __restrict__ Bl,
                      const float* __restrict__ nA, const float* __restrict__ nB,
                      float* __restrict__ distOut, int* __restrict__ predOut) {
  __shared__ bf16 sAh[2][16 * C_DIM];
  __shared__ bf16 sAl[2][16 * C_DIM];
  __shared__ float sD[2][16][17];

  const int lane = threadIdx.x & 31;
  const int wave = threadIdx.x >> 5;
  const int ibase = blockIdx.x * 32 + wave * 16;

  // Stage this wave's 16 A rows (hi+lo) into LDS with 128-bit copies.
  {
    const v8bf* gh = (const v8bf*)(Ah + (size_t)ibase * C_DIM);
    const v8bf* gl = (const v8bf*)(Al + (size_t)ibase * C_DIM);
    v8bf* lh = (v8bf*)sAh[wave];
    v8bf* ll = (v8bf*)sAl[wave];
    for (int c = lane; c < (16 * C_DIM) / 8; c += 32) { lh[c] = gh[c]; ll[c] = gl[c]; }
  }
  __syncthreads();

  const int m  = lane & 15;   // A row within tile / B column within tile
  const int kh = lane >> 4;   // K-half selector (contiguous k = 16*kh + s for A and B)

  float d0 = 3.4e38f, d1 = 3.4e38f, d2 = 3.4e38f;
  int   i0 = 0, i1 = 0, i2 = 0;
  const float nAi = nA[ibase + m];  // used by lanes 0..15 in scan phase

  const bf16* browh = Bh + (size_t)m * C_DIM + kh * 16;
  const bf16* browl = Bl + (size_t)m * C_DIM + kh * 16;

  for (int jt = 0; jt < L_TOT / 16; ++jt) {
    const int jbase = jt * 16;
    v8f accHH = {}; v8f accHL = {}; v8f accLH = {};
    const size_t boff = (size_t)jbase * C_DIM;

    for (int kb = 0; kb < C_DIM / 32; ++kb) {
      const int kbase = kb * 32;
      // A fragment (16x32 bf16) from LDS: two 8-element runs per lane.
      V16 a_h, a_l;
      const v8bf* ah8 = (const v8bf*)&sAh[wave][m * C_DIM + kbase];
      const v8bf* al8 = (const v8bf*)&sAl[wave][m * C_DIM + kbase];
      a_h.h[0] = ah8[2 * kh]; a_h.h[1] = ah8[2 * kh + 1];
      a_l.h[0] = al8[2 * kh]; a_l.h[1] = al8[2 * kh + 1];
      // B fragment (32x16 bf16 = f1 tile^T): 16 contiguous K per lane.
      V16 b_h, b_l;
      const v8bf* bh8 = (const v8bf*)(browh + boff + kbase);
      const v8bf* bl8 = (const v8bf*)(browl + boff + kbase);
      b_h.h[0] = bh8[0]; b_h.h[1] = bh8[1];
      b_l.h[0] = bl8[0]; b_l.h[1] = bl8[1];
      // Split products: hi*hi + hi*lo + lo*hi (independent accumulators).
      accHH = __builtin_amdgcn_wmma_f32_16x16x32_bf16(false, a_h.v, false, b_h.v,
                                                      (short)0, accHH, false, false);
      accHL = __builtin_amdgcn_wmma_f32_16x16x32_bf16(false, a_h.v, false, b_l.v,
                                                      (short)0, accHL, false, false);
      accLH = __builtin_amdgcn_wmma_f32_16x16x32_bf16(false, a_l.v, false, b_h.v,
                                                      (short)0, accLH, false, false);
    }

    __syncthreads();  // previous scan of sD finished
    // C/D layout: lane holds column N=m, rows r + 8*kh.
    #pragma unroll
    for (int r = 0; r < 8; ++r)
      sD[wave][r + 8 * kh][m] = accHH[r] + accHL[r] + accLH[r];
    __syncthreads();

    if (lane < 16) {
      #pragma unroll
      for (int jj = 0; jj < 16; ++jj) {
        const int j = jbase + jj;
        const float dv = nAi + nB[j] - 2.f * sD[wave][lane][jj];
        if (dv < d0)      { d2 = d1; i2 = i1; d1 = d0; i1 = i0; d0 = dv; i0 = j; }
        else if (dv < d1) { d2 = d1; i2 = i1; d1 = dv; i1 = j; }
        else if (dv < d2) { d2 = dv; i2 = j; }
      }
    }
  }

  if (lane < 16) {
    const int i = ibase + lane;
    distOut[i * 3 + 0] = d0; distOut[i * 3 + 1] = d1; distOut[i * 3 + 2] = d2;
    predOut[i * 3 + 0] = i0; predOut[i * 3 + 1] = i1; predOut[i * 3 + 2] = i2;
  }
}

// ---------------------------------------------------------------------------
// Epilogue helpers
// ---------------------------------------------------------------------------
__device__ __forceinline__ bool border_ok(int i) {
  int x = i % LSIDE, y = i / LSIDE;
  return x >= 2 && x < LSIDE - 2 && y >= 2 && y < LSIDE - 2;
}

__device__ __forceinline__ float theta_f(float ax, float ay, float bx, float by,
                                         float cx, float cy) {
  float abx = ax - bx, aby = ay - by, acx = ax - cx, acy = ay - cy;
  float denom = fmaxf(hypotf(abx, aby) * hypotf(acx, acy), 1e-8f);
  float cosv = (abx * acx + aby * acy) / denom;
  cosv = fminf(fmaxf(cosv, -0.99f), 0.99f);
  return acosf(cosv);
}

// Output layout (floats), flat in reference tuple order:
//  [0,3L) sqrt(dist1) | [3L,6L) preds1 | [6L,9L) sqrt(dist2) | [9L,12L) preds2
//  [12L,13L) mutual | [13L,14L) dist_means | [14L,15L) angle_means | [15L,16L) geom_keep
__global__ void epi1_kernel(const float* __restrict__ dist1, const int* __restrict__ preds1,
                            const float* __restrict__ dist2, const int* __restrict__ preds2,
                            const int* __restrict__ sidx,
                            float* __restrict__ ratioRow, float* __restrict__ rsum,
                            float* __restrict__ out) {
  int i = blockIdx.x * blockDim.x + threadIdx.x;
  if (i >= L_TOT) return;

  bool keep1 = (dist1[i * 3 + 0] / dist1[i * 3 + 1] <= 0.85f) && border_ok(i);
  int  j0    = preds1[i * 3 + 0];
  bool keep2 = (dist2[j0 * 3 + 0] / dist2[j0 * 3 + 1] <= 0.85f) && border_ok(j0);
  bool mutual = keep1 && keep2 && (preds2[j0 * 3 + 0] == i);
  out[12 * L_TOT + i] = mutual ? 1.f : 0.f;

  float x1i = (float)(i % LSIDE),  y1i = (float)(i / LSIDE);
  float x2i = (float)(j0 % LSIDE), y2i = (float)(j0 / LSIDE);

  float x1s[5], y1s[5], x2s[5], y2s[5];
  float rs = 0.f;
  for (int s = 0; s < 5; ++s) {
    int t = sidx[i * 5 + s];
    x1s[s] = (float)(t % LSIDE); y1s[s] = (float)(t / LSIDE);
    int tj = preds1[t * 3 + 0];
    x2s[s] = (float)(tj % LSIDE); y2s[s] = (float)(tj / LSIDE);
    float d1sv = hypotf(x1i - x1s[s], y1i - y1s[s]);
    float d2sv = hypotf(x2i - x2s[s], y2i - y2s[s]);
    float rr = d1sv / fmaxf(d2sv, 1e-6f);
    ratioRow[i * 5 + s] = rr;
    rs += rr;
  }
  rsum[i] = mutual ? rs : 0.f;

  const int bsel[5] = {4, 1, 2, 3, 4};
  const int csel[5] = {0, 0, 1, 2, 3};
  float am = 0.f;
  for (int s = 0; s < 5; ++s) {
    int bs = bsel[s], cs = csel[s];
    float t1 = theta_f(x1i, y1i, x1s[bs], y1s[bs], x1s[cs], y1s[cs]);
    float t2 = theta_f(x2i, y2i, x2s[bs], y2s[bs], x2s[cs], y2s[cs]);
    float a1 = (t1 + 0.1f) * 57.29577951308232f;
    float a2 = (t2 + 0.1f) * 57.29577951308232f;
    am += fabsf(a1 / a2 - 1.f);
  }
  out[14 * L_TOT + i] = am * 0.2f;

  for (int k = 0; k < 3; ++k) {
    out[i * 3 + k]             = sqrtf(dist1[i * 3 + k]);
    out[3 * L_TOT + i * 3 + k] = (float)preds1[i * 3 + k];
    out[6 * L_TOT + i * 3 + k] = sqrtf(dist2[i * 3 + k]);
    out[9 * L_TOT + i * 3 + k] = (float)preds2[i * 3 + k];
  }
}

// Deterministic fixed-order reduction (no float atomics -> replay-stable).
__global__ void reduce_kernel(const float* __restrict__ rsum,
                              const float* __restrict__ mflag,
                              float* __restrict__ accum) {
  __shared__ float s0[256], s1[256];
  int t = threadIdx.x;
  float a = 0.f, b = 0.f;
  for (int i = t; i < L_TOT; i += 256) { a += rsum[i]; b += mflag[i]; }
  s0[t] = a; s1[t] = b; __syncthreads();
  for (int off = 128; off > 0; off >>= 1) {
    if (t < off) { s0[t] += s0[t + off]; s1[t] += s1[t + off]; }
    __syncthreads();
  }
  if (t == 0) { accum[0] = s0[0]; accum[1] = s1[0]; }
}

__global__ void epi2_kernel(const float* __restrict__ ratioRow,
                            const float* __restrict__ accum,
                            float* __restrict__ out) {
  int i = blockIdx.x * blockDim.x + threadIdx.x;
  if (i >= L_TOT) return;
  float cnt = fmaxf(accum[1], 1.f);
  float dc  = accum[0] / (cnt * 5.f);
  float dm = 0.f;
  for (int s = 0; s < 5; ++s) dm += fabsf(ratioRow[i * 5 + s] / dc - 1.f);
  dm *= 0.2f;
  out[13 * L_TOT + i] = dm;
  bool mut = out[12 * L_TOT + i] > 0.5f;
  float am = out[14 * L_TOT + i];
  out[15 * L_TOT + i] = (mut && dm < 0.1f && am < 0.1f) ? 1.f : 0.f;
}

// ---------------------------------------------------------------------------
// Launch. Workspace use ~13.7 MB (bf16 hi/lo copies, norms, knn results,
// ratio rows, reduction scratch).
// ---------------------------------------------------------------------------
extern "C" void kernel_launch(void* const* d_in, const int* in_sizes, int n_in,
                              void* d_out, int out_size, void* d_ws, size_t ws_size,
                              hipStream_t stream) {
  const float* feat0 = (const float*)d_in[0];
  const float* feat1 = (const float*)d_in[1];
  const int*   sidx  = (const int*)d_in[2];
  float* out = (float*)d_out;

  char* w = (char*)d_ws;
  const size_t LC = (size_t)L_TOT * C_DIM;
  bf16* f0h = (bf16*)w; w += LC * sizeof(bf16);
  bf16* f0l = (bf16*)w; w += LC * sizeof(bf16);
  bf16* f1h = (bf16*)w; w += LC * sizeof(bf16);
  bf16* f1l = (bf16*)w; w += LC * sizeof(bf16);
  float* n0 = (float*)w; w += L_TOT * sizeof(float);
  float* n1 = (float*)w; w += L_TOT * sizeof(float);
  float* dist1 = (float*)w; w += L_TOT * 3 * sizeof(float);
  int*   preds1 = (int*)w;  w += L_TOT * 3 * sizeof(int);
  float* dist2 = (float*)w; w += L_TOT * 3 * sizeof(float);
  int*   preds2 = (int*)w;  w += L_TOT * 3 * sizeof(int);
  float* ratioRow = (float*)w; w += L_TOT * 5 * sizeof(float);
  float* rsum = (float*)w;     w += L_TOT * sizeof(float);
  float* accum = (float*)w;    w += 2 * sizeof(float);

  prep_kernel<<<(2 * L_TOT + 255) / 256, 256, 0, stream>>>(
      feat0, feat1, f0h, f0l, f1h, f1l, n0, n1);

  // Row pass: top-3 of d over j.
  gemm_top3_kernel<<<L_TOT / 32, 64, 0, stream>>>(
      f0h, f0l, f1h, f1l, n0, n1, dist1, preds1);
  // Column pass: top-3 of d^T over i (role swap; d form is symmetric).
  gemm_top3_kernel<<<L_TOT / 32, 64, 0, stream>>>(
      f1h, f1l, f0h, f0l, n1, n0, dist2, preds2);

  epi1_kernel<<<(L_TOT + 255) / 256, 256, 0, stream>>>(
      dist1, preds1, dist2, preds2, sidx, ratioRow, rsum, out);
  reduce_kernel<<<1, 256, 0, stream>>>(rsum, out + 12 * L_TOT, accum);
  epi2_kernel<<<(L_TOT + 255) / 256, 256, 0, stream>>>(ratioRow, accum, out);
}
